// GAE_GMNN_RGCN_DP_delta_13846974562756
// MI455X (gfx1250) — matrix operand order repaired and checked by reading, not compiled, not run on hardware
//
#include <hip/hip_runtime.h>

typedef __attribute__((ext_vector_type(16))) __bf16 v16bf;
typedef __attribute__((ext_vector_type(8)))  float  v8f;

#define NP 10000
#define NJ 10000
#define NB 10000
#define NA 20000
#define NTOT 50000
#define EHET 160000
#define EHOMO 160000

__device__ __forceinline__ unsigned short f2bf(float f) {
  unsigned int u = __float_as_uint(f);
  unsigned int r = u + 0x7FFFu + ((u >> 16) & 1u);          // round to nearest even
  if ((u & 0x7FFFFFFFu) > 0x7F800000u) r = u | 0x00400000u; // keep NaN quiet
  return (unsigned short)(r >> 16);
}

__device__ __forceinline__ int node_type(int i) {
  return (i < NP) ? 0 : (i < NP + NJ) ? 1 : (i < NP + NJ + NB) ? 2 : 3;
}

// ---------------------------------------------------------------------------
// Fragment-order bf16 packing.
// A (16x32 tile): lane(half,m) element j <-> K-in-chunk c = 16*half + j, so
//   p(c) = c (same as B); lane reads 32 contiguous bytes.
// Ap layout: [K/32][M][32] ushort.
// ---------------------------------------------------------------------------
__global__ void pack_a(const float* __restrict__ A, const float* __restrict__ scale,
                       unsigned short* __restrict__ Ap, int M, int K, int lda)
{
  long long t = (long long)blockIdx.x * blockDim.x + threadIdx.x;
  if (t >= (long long)M * K) return;
  int row = (int)(t / K), k = (int)(t % K);
  float v = A[(size_t)row * lda + k];
  if (scale) v *= scale[row];
  int c = k & 31;
  int p = c;
  Ap[((size_t)(k >> 5) * M + row) * 32 + p] = f2bf(v);
}

// B (32x16 tile): lane(half,n) element j <-> K-in-chunk c = 16*half + j, so p = c.
// Bp layout: [K/32][N][32] ushort.
__global__ void pack_b(const float* __restrict__ B, unsigned short* __restrict__ Bp,
                       int K, int N)
{
  long long t = (long long)blockIdx.x * blockDim.x + threadIdx.x;
  if (t >= (long long)K * N) return;
  int k = (int)(t / N), n = (int)(t % N);
  Bp[((size_t)(k >> 5) * N + n) * 32 + (k & 31)] = f2bf(B[t]);
}

// ---------------------------------------------------------------------------
// WMMA GEMM on packed operands: C[M,*](ldc,colOff) (+)= A @ B.
// Block = 128 threads (4 waves); block tile = 16 rows x 128 cols; each wave
// owns two 16x16 col tiles -> 2 v_wmma per 32-K step, pure b128 loads, no LDS.
// Requires: M % 16 == 0, K % 32 == 0, grid.y * 128 == N (true at all call sites).
// ---------------------------------------------------------------------------
__global__ __launch_bounds__(128) void gemm_wmma_bf16(
    const unsigned short* __restrict__ Ap, const unsigned short* __restrict__ Bp,
    float* __restrict__ C, int M, int N, int K, int ldc, int colOff, int beta)
{
  const int wave = threadIdx.x >> 5;
  const int lane = threadIdx.x & 31;
  const int half = lane >> 4;
  const int nrow = lane & 15;
  const int rowBase = blockIdx.x * 16;
  const int colBase = blockIdx.y * 128 + wave * 32;

  v8f acc0 = {0.f, 0.f, 0.f, 0.f, 0.f, 0.f, 0.f, 0.f};
  v8f acc1 = acc0;
  if (beta) {
#pragma unroll
    for (int i = 0; i < 8; ++i) {
      int r = rowBase + half * 8 + i;
      acc0[i] = C[(size_t)r * ldc + colOff + colBase + nrow];
      acc1[i] = C[(size_t)r * ldc + colOff + colBase + 16 + nrow];
    }
  }

  const unsigned short* ap  = Ap + ((size_t)(rowBase + nrow) * 32) + half * 16;
  const unsigned short* bp0 = Bp + ((size_t)(colBase + nrow) * 32) + half * 16;
  const unsigned short* bp1 = bp0 + 16 * 32;
  const size_t aStep = (size_t)M * 32;
  const size_t bStep = (size_t)N * 32;

  const int nk = K >> 5;
  for (int kc = 0; kc < nk; ++kc) {
    union { v16bf v; uint4 q[2]; } a, b0, b1;
    a.q[0]  = *(const uint4*)(ap);
    a.q[1]  = *(const uint4*)(ap + 8);
    b0.q[0] = *(const uint4*)(bp0);
    b0.q[1] = *(const uint4*)(bp0 + 8);
    b1.q[0] = *(const uint4*)(bp1);
    b1.q[1] = *(const uint4*)(bp1 + 8);
    acc0 = __builtin_amdgcn_wmma_f32_16x16x32_bf16(false, a.v, false, b0.v,
                                                   (short)0, acc0, false, false);
    acc1 = __builtin_amdgcn_wmma_f32_16x16x32_bf16(false, a.v, false, b1.v,
                                                   (short)0, acc1, false, false);
    ap += aStep; bp0 += bStep; bp1 += bStep;
  }

#pragma unroll
  for (int i = 0; i < 8; ++i) {
    int r = rowBase + half * 8 + i;
    C[(size_t)r * ldc + colOff + colBase + nrow]      = acc0[i];
    C[(size_t)r * ldc + colOff + colBase + 16 + nrow] = acc1[i];
  }
}

// ------------------- feature assembly: X[i, 0:128] ------------------------
__global__ void build_x(const float* __restrict__ fp, const float* __restrict__ fj,
                        const float* __restrict__ fb, const float* __restrict__ fa,
                        const float* __restrict__ gm, const float* __restrict__ dpc,
                        const float* __restrict__ de, const float* __restrict__ dj,
                        const float* __restrict__ db, float* __restrict__ X)
{
  long long tid = (long long)blockIdx.x * blockDim.x + threadIdx.x;
  if (tid >= (long long)NTOT * 128) return;
  int i = (int)(tid >> 7), c = (int)(tid & 127);
  int li; const float* f; const float* dp;
  if (i < NP)            { li = i;                f = fp; dp = de; }
  else if (i < NP + NJ)  { li = i - NP;           f = fj; dp = dj; }
  else if (i < NP+NJ+NB) { li = i - NP - NJ;      f = fb; dp = db; }
  else                   { li = i - NP - NJ - NB; f = fa; dp = nullptr; }
  float v;
  if (c < 110)       v = f[(size_t)li * 110 + c];
  else if (c < 126)  v = gm[(size_t)i * 16 + (c - 110)];
  else if (c == 126) v = dpc[i];
  else               v = dp ? dp[li] : 0.f;
  X[tid] = v;
}

// ------------------------------ small utilities ----------------------------
__global__ void fill_f(float* __restrict__ a, float v, long long n) {
  long long t = (long long)blockIdx.x * blockDim.x + threadIdx.x;
  if (t < n) a[t] = v;
}

__global__ void relu_k(float* __restrict__ a, long long n) {
  long long t = (long long)blockIdx.x * blockDim.x + threadIdx.x;
  if (t < n) a[t] = fmaxf(a[t], 0.f);
}

__global__ void count_deg(const int* __restrict__ idx, float* __restrict__ deg, int E) {
  int e = blockIdx.x * blockDim.x + threadIdx.x;
  if (e < E) atomicAdd(&deg[idx[e]], 1.0f);
}

__global__ void inv_sqrt_deg(float* __restrict__ a, int n) {
  int i = blockIdx.x * blockDim.x + threadIdx.x;
  if (i < n) a[i] = (a[i] > 0.f) ? rsqrtf(a[i]) : 0.f;
}

__global__ void mean_div(float* __restrict__ M, const float* __restrict__ deg,
                         int n, int F) {
  long long t = (long long)blockIdx.x * blockDim.x + threadIdx.x;
  if (t >= (long long)n * F) return;
  M[t] = M[t] / fmaxf(deg[(int)(t / F)], 1.f);
}

__global__ void bias_init(float* __restrict__ Out, const float* __restrict__ b,
                          int rows, int ldc, int colOff, int F) {
  long long t = (long long)blockIdx.x * blockDim.x + threadIdx.x;
  if (t >= (long long)rows * F) return;
  int i = (int)(t / F), c = (int)(t % F);
  Out[(size_t)i * ldc + colOff + c] = b[c];
}

__global__ void fill_block(float* __restrict__ Out, int rows, int ldc, int colOff,
                           int F, float v) {
  long long t = (long long)blockIdx.x * blockDim.x + threadIdx.x;
  if (t >= (long long)rows * F) return;
  int i = (int)(t / F), c = (int)(t % F);
  Out[(size_t)i * ldc + colOff + c] = v;
}

// combined per-dst-type bias (sum over relations targeting each type)
__global__ void combine_bias(const float* __restrict__ b, float* __restrict__ bc, int F) {
  int t = blockIdx.x * blockDim.x + threadIdx.x;
  if (t >= 4 * F) return;
  int ty = t / F, c = t % F;
  const int dt[10] = {0, 1, 0, 1, 0, 2, 3, 2, 3, 1};
  float s = 0.f;
  for (int r = 0; r < 10; ++r)
    if (dt[r] == ty) s += b[r * F + c];
  bc[t] = s;
}

__global__ void het_bias_init(float* __restrict__ Out, const float* __restrict__ bc,
                              int ldc, int colOff, int F) {
  long long t = (long long)blockIdx.x * blockDim.x + threadIdx.x;
  if (t >= (long long)NTOT * F) return;
  int i = (int)(t / F), c = (int)(t % F);
  Out[(size_t)i * ldc + colOff + c] = bc[node_type(i) * F + c];
}

// ------------------- edge scatters (segment-sum with D^-1/2 folded) -------
__global__ void het_scatter256(const int* __restrict__ src, const int* __restrict__ dst,
                               const float* __restrict__ Y, const float* __restrict__ ndv,
                               float* __restrict__ Out, int dstOff, int ldo, int colOff)
{
  long long tid = (long long)blockIdx.x * blockDim.x + threadIdx.x;
  int e = (int)(tid >> 6);
  if (e >= EHET) return;
  int c4 = ((int)tid & 63) << 2;
  int s = src[e], d = dst[e];
  float w = ndv[d];
  const float4 y = *(const float4*)(Y + (size_t)s * 256 + c4);
  float* o = Out + (size_t)(dstOff + d) * ldo + colOff + c4;
  atomicAdd(o + 0, y.x * w);
  atomicAdd(o + 1, y.y * w);
  atomicAdd(o + 2, y.z * w);
  atomicAdd(o + 3, y.w * w);
}

__global__ void het_scatter2(const int* __restrict__ src, const int* __restrict__ dst,
                             const float* __restrict__ Y, const float* __restrict__ ndv,
                             float* __restrict__ Out, int dstOff)
{
  int e = blockIdx.x * blockDim.x + threadIdx.x;
  if (e >= EHET) return;
  int s = src[e], d = dst[e];
  float w = ndv[d];
  atomicAdd(&Out[(size_t)(dstOff + d) * 2 + 0], Y[(size_t)s * 2 + 0] * w);
  atomicAdd(&Out[(size_t)(dstOff + d) * 2 + 1], Y[(size_t)s * 2 + 1] * w);
}

__global__ void sage_scatter(const int* __restrict__ src, const int* __restrict__ dst,
                             const float* __restrict__ X, int lda, int F,
                             float* __restrict__ M)
{
  long long tid = (long long)blockIdx.x * blockDim.x + threadIdx.x;
  int ch = F >> 2;
  int e = (int)(tid / ch);
  if (e >= EHOMO) return;
  int c4 = (int)(tid % ch) << 2;
  int s = src[e], d = dst[e];
  const float4 x = *(const float4*)(X + (size_t)s * lda + c4);
  float* o = M + (size_t)d * F + c4;
  atomicAdd(o + 0, x.x);
  atomicAdd(o + 1, x.y);
  atomicAdd(o + 2, x.z);
  atomicAdd(o + 3, x.w);
}

// tiny output head: Y[M,2] = (A ⊙ scale) @ B[K,2]
__global__ void gemm_small2(const float* __restrict__ A, const float* __restrict__ scale,
                            const float* __restrict__ B, float* __restrict__ Y,
                            int M, int K, int lda)
{
  int i = blockIdx.x * blockDim.x + threadIdx.x;
  if (i >= M) return;
  float sc = scale ? scale[i] : 1.f;
  float s0 = 0.f, s1 = 0.f;
  const float* a = A + (size_t)i * lda;
  for (int k = 0; k < K; ++k) {
    float v = a[k];
    s0 += v * B[2 * k];
    s1 += v * B[2 * k + 1];
  }
  Y[(size_t)i * 2 + 0] = s0 * sc;
  Y[(size_t)i * 2 + 1] = s1 * sc;
}

__global__ void softmax2(const float* __restrict__ Xv, float* __restrict__ O, int n) {
  int i = blockIdx.x * blockDim.x + threadIdx.x;
  if (i >= n) return;
  float a = Xv[2 * i], b = Xv[2 * i + 1];
  float m = fmaxf(a, b);
  float ea = __expf(a - m), eb = __expf(b - m);
  float s = ea + eb;
  O[2 * i]     = ea / s;
  O[2 * i + 1] = eb / s;
}

// ===========================================================================
extern "C" void kernel_launch(void* const* d_in, const int* in_sizes, int n_in,
                              void* d_out, int out_size, void* d_ws, size_t ws_size,
                              hipStream_t stream)
{
  (void)in_sizes; (void)n_in; (void)out_size; (void)ws_size;
  const float* feat_power = (const float*)d_in[0];
  const float* feat_junc  = (const float*)d_in[1];
  const float* feat_bs    = (const float*)d_in[2];
  const float* feat_aoi   = (const float*)d_in[3];
  const float* gmnn       = (const float*)d_in[4];
  const float* dpc        = (const float*)d_in[5];
  const float* dpe        = (const float*)d_in[6];
  const float* dpj        = (const float*)d_in[7];
  const float* dpb        = (const float*)d_in[8];
  const int*   het_src    = (const int*)d_in[9];
  const int*   het_dst    = (const int*)d_in[10];
  const int*   homo_src   = (const int*)d_in[11];
  const int*   homo_dst   = (const int*)d_in[12];
  const float* ce1_W = (const float*)d_in[13]; const float* ce1_b = (const float*)d_in[14];
  const float* ce2_W = (const float*)d_in[15]; const float* ce2_b = (const float*)d_in[16];
  const float* l1a_W = (const float*)d_in[17]; const float* l1a_b = (const float*)d_in[18];
  const float* l1b_W = (const float*)d_in[19]; const float* l1b_b = (const float*)d_in[20];
  const float* cd1_W = (const float*)d_in[21]; const float* cd1_b = (const float*)d_in[22];
  const float* cd2_W = (const float*)d_in[23]; const float* cd2_b = (const float*)d_in[24];
  const float* l2a_W = (const float*)d_in[25]; const float* l2a_b = (const float*)d_in[26];
  const float* l2b_W = (const float*)d_in[27]; const float* l2b_b = (const float*)d_in[28];
  const float* l3a_W = (const float*)d_in[29]; const float* l3a_b = (const float*)d_in[30];
  const float* l3b_W = (const float*)d_in[31]; const float* l3b_b = (const float*)d_in[32];
  const float* enc_W1s = (const float*)d_in[33]; const float* enc_W1n = (const float*)d_in[34];
  const float* enc_b1  = (const float*)d_in[35];
  const float* enc_W2s = (const float*)d_in[36]; const float* enc_W2n = (const float*)d_in[37];
  const float* enc_b2  = (const float*)d_in[38];
  const float* dec_W1s = (const float*)d_in[39]; const float* dec_W1n = (const float*)d_in[40];
  const float* dec_b1  = (const float*)d_in[41];
  const float* dec_W2s = (const float*)d_in[42]; const float* dec_W2n = (const float*)d_in[43];
  const float* dec_b2  = (const float*)d_in[44];
  float* out = (float*)d_out;

  // ----- workspace carve (bump allocator, 256B aligned) -----
  char* wp = (char*)d_ws;
  auto wsalloc = [&](size_t bytes) -> void* {
    void* r = (void*)wp;
    wp += (bytes + 255) & ~(size_t)255;
    return r;
  };
  float* X    = (float*)wsalloc((size_t)NTOT * 128 * 4);   // augmented features
  float* H    = (float*)wsalloc((size_t)NTOT * 256 * 4);   // hetero hidden
  float* Z512 = (float*)wsalloc((size_t)NTOT * 512 * 4);   // concat [rgcn | sage]
  float* Z2   = (float*)wsalloc((size_t)NTOT * 256 * 4);   // 256-wide latent
  float* Yb   = (float*)wsalloc((size_t)20000 * 256 * 4);  // per-relation GEMM scratch
  float* T1   = (float*)wsalloc((size_t)10000 * 256 * 4);  // SAGE layer-1 hidden
  float* Mb   = (float*)wsalloc((size_t)10000 * 256 * 4);  // SAGE neighbor mean
  float* nsA  = (float*)wsalloc((size_t)10 * 20000 * 4);   // D_src^-1/2 per relation
  float* ndA  = (float*)wsalloc((size_t)10 * 20000 * 4);   // D_dst^-1/2 per relation
  float* hdeg = (float*)wsalloc((size_t)3 * 10000 * 4);    // homo in-degree (counts)
  float* bC   = (float*)wsalloc((size_t)4 * 256 * 4);      // combined hetero bias
  float* L1   = (float*)wsalloc((size_t)NTOT * 2 * 4);
  float* L2   = (float*)wsalloc((size_t)NTOT * 2 * 4);
  unsigned short* ApW = (unsigned short*)wsalloc((size_t)20000 * 512 * 2); // packed A
  unsigned short* BpW = (unsigned short*)wsalloc((size_t)512 * 256 * 2);   // packed B

  static const int ST[10]  = {0, 0, 1, 1, 2, 0, 2, 3, 1, 3};
  static const int DTt[10] = {0, 1, 0, 1, 0, 2, 3, 2, 3, 1};
  static const int OFF[4]  = {0, NP, NP + NJ, NP + NJ + NB};
  static const int CNT[4]  = {NP, NJ, NB, NA};

  auto nb = [](long long n) { return (unsigned)((n + 255) / 256); };

  // packed-operand GEMM: C[M,*](ldc,colOff) (+)= (A ⊙ scale) @ B   (N == 256)
  auto gemm = [&](const float* A, const float* scale, int lda,
                  const float* B, float* C, int M, int K,
                  int ldc, int colOff, int beta) {
    pack_a<<<nb((long long)M * K), 256, 0, stream>>>(A, scale, ApW, M, K, lda);
    pack_b<<<nb((long long)K * 256), 256, 0, stream>>>(B, BpW, K, 256);
    dim3 g(M / 16, 256 / 128);
    gemm_wmma_bf16<<<g, 128, 0, stream>>>(ApW, BpW, C, M, 256, K, ldc, colOff, beta);
  };

  // ----- 1) assemble X -----
  build_x<<<nb((long long)NTOT * 128), 256, 0, stream>>>(
      feat_power, feat_junc, feat_bs, feat_aoi, gmnn, dpc, dpe, dpj, dpb, X);

  // ----- 2) degrees (identical for every layer -> computed once) -----
  fill_f<<<nb(10LL * 20000), 256, 0, stream>>>(nsA, 0.f, 10LL * 20000);
  fill_f<<<nb(10LL * 20000), 256, 0, stream>>>(ndA, 0.f, 10LL * 20000);
  fill_f<<<nb(3LL * 10000), 256, 0, stream>>>(hdeg, 0.f, 3LL * 10000);
  for (int r = 0; r < 10; ++r) {
    count_deg<<<nb(EHET), 256, 0, stream>>>(het_src + (size_t)r * EHET, nsA + r * 20000, EHET);
    count_deg<<<nb(EHET), 256, 0, stream>>>(het_dst + (size_t)r * EHET, ndA + r * 20000, EHET);
  }
  for (int r = 0; r < 10; ++r) {
    inv_sqrt_deg<<<nb(CNT[ST[r]]), 256, 0, stream>>>(nsA + r * 20000, CNT[ST[r]]);
    inv_sqrt_deg<<<nb(CNT[DTt[r]]), 256, 0, stream>>>(ndA + r * 20000, CNT[DTt[r]]);
  }
  for (int g = 0; g < 3; ++g)
    count_deg<<<nb(EHOMO), 256, 0, stream>>>(homo_dst + (size_t)g * EHOMO, hdeg + g * 10000, EHOMO);

  // ----- hetero GraphConv (Fout = 256) -----
  auto hconv = [&](const float* Xin, int ldx, int Fin,
                   const float* W, const float* b, float* Out, int ldo, int colOff) {
    combine_bias<<<nb(4 * 256), 256, 0, stream>>>(b, bC, 256);
    het_bias_init<<<nb((long long)NTOT * 256), 256, 0, stream>>>(Out, bC, ldo, colOff, 256);
    for (int r = 0; r < 10; ++r) {
      int st = ST[r], dt = DTt[r];
      int m = CNT[st];
      gemm(Xin + (size_t)OFF[st] * ldx, nsA + r * 20000, ldx,
           W + (size_t)r * Fin * 256, Yb, m, Fin, 256, 0, 0);
      het_scatter256<<<nb((long long)EHET * 64), 256, 0, stream>>>(
          het_src + (size_t)r * EHET, het_dst + (size_t)r * EHET,
          Yb, ndA + r * 20000, Out, OFF[dt], ldo, colOff);
    }
  };

  // ----- hetero GraphConv (Fout = 2, scalar path) -----
  auto hconv2 = [&](const float* Xin, int ldx, int Fin,
                    const float* W, const float* b, float* Out) {
    combine_bias<<<nb(4 * 2), 256, 0, stream>>>(b, bC, 2);
    het_bias_init<<<nb((long long)NTOT * 2), 256, 0, stream>>>(Out, bC, 2, 0, 2);
    for (int r = 0; r < 10; ++r) {
      int st = ST[r], dt = DTt[r];
      int m = CNT[st];
      gemm_small2<<<nb(m), 256, 0, stream>>>(Xin + (size_t)OFF[st] * ldx, nsA + r * 20000,
                                             W + (size_t)r * Fin * 2, Yb, m, Fin, ldx);
      het_scatter2<<<nb(EHET), 256, 0, stream>>>(het_src + (size_t)r * EHET,
                                                 het_dst + (size_t)r * EHET,
                                                 Yb, ndA + r * 20000, Out, OFF[dt]);
    }
  };

  // ----- two-layer SAGE (mean aggregator), writes cols [colOff, colOff+256) -----
  auto sage2 = [&](int g, const float* Xg, int Fin,
                   const float* W1s, const float* W1n, const float* b1,
                   const float* W2s, const float* W2n, const float* b2,
                   float* OutBase, int ldo, int colOff) {
    const int n = 10000;
    const int* s = homo_src + (size_t)g * EHOMO;
    const int* d = homo_dst + (size_t)g * EHOMO;
    const float* dg = hdeg + g * 10000;
    // layer 1
    fill_f<<<nb((long long)n * Fin), 256, 0, stream>>>(Mb, 0.f, (long long)n * Fin);
    sage_scatter<<<nb((long long)EHOMO * (Fin / 4)), 256, 0, stream>>>(s, d, Xg, Fin, Fin, Mb);
    mean_div<<<nb((long long)n * Fin), 256, 0, stream>>>(Mb, dg, n, Fin);
    bias_init<<<nb((long long)n * 256), 256, 0, stream>>>(T1, b1, n, 256, 0, 256);
    gemm(Xg, nullptr, Fin, W1s, T1, n, Fin, 256, 0, 1);
    gemm(Mb, nullptr, Fin, W1n, T1, n, Fin, 256, 0, 1);
    relu_k<<<nb((long long)n * 256), 256, 0, stream>>>(T1, (long long)n * 256);
    // layer 2
    fill_f<<<nb((long long)n * 256), 256, 0, stream>>>(Mb, 0.f, (long long)n * 256);
    sage_scatter<<<nb((long long)EHOMO * 64), 256, 0, stream>>>(s, d, T1, 256, 256, Mb);
    mean_div<<<nb((long long)n * 256), 256, 0, stream>>>(Mb, dg, n, 256);
    float* Cb = OutBase + (size_t)OFF[g] * ldo;
    bias_init<<<nb((long long)n * 256), 256, 0, stream>>>(Cb, b2, n, ldo, colOff, 256);
    gemm(T1, nullptr, 256, W2s, Cb, n, 256, ldo, colOff, 1);
    gemm(Mb, nullptr, 256, W2n, Cb, n, 256, ldo, colOff, 1);
  };

  // ----- 3) CoupledEncoder RGCN -----
  hconv(X, 128, 128, ce1_W, ce1_b, H, 256, 0);
  relu_k<<<nb((long long)NTOT * 256), 256, 0, stream>>>(H, (long long)NTOT * 256);
  hconv(H, 256, 256, ce2_W, ce2_b, Z512, 512, 0);

  // ----- 4) per-type SAGE encoders into cols 256..511 -----
  for (int g = 0; g < 3; ++g)
    sage2(g, X + (size_t)OFF[g] * 128, 128,
          enc_W1s + (size_t)g * 128 * 256, enc_W1n + (size_t)g * 128 * 256, enc_b1 + g * 256,
          enc_W2s + (size_t)g * 256 * 256, enc_W2n + (size_t)g * 256 * 256, enc_b2 + g * 256,
          Z512, 512, 256);
  fill_block<<<nb((long long)NA * 256), 256, 0, stream>>>(
      Z512 + (size_t)OFF[3] * 512, NA, 512, 256, 256, 0.f);

  // ----- 5) latent RGCN (l1) -> Z2 -----
  hconv(Z512, 512, 512, l1a_W, l1a_b, H, 256, 0);
  relu_k<<<nb((long long)NTOT * 256), 256, 0, stream>>>(H, (long long)NTOT * 256);
  hconv(H, 256, 256, l1b_W, l1b_b, Z2, 256, 0);

  // ----- 6) CoupledDecoder RGCN -> Z512 cols 0..255 -----
  hconv(Z2, 256, 256, cd1_W, cd1_b, H, 256, 0);
  relu_k<<<nb((long long)NTOT * 256), 256, 0, stream>>>(H, (long long)NTOT * 256);
  hconv(H, 256, 256, cd2_W, cd2_b, Z512, 512, 0);

  // ----- 7) per-type SAGE decoders into cols 256..511 -----
  for (int g = 0; g < 3; ++g)
    sage2(g, Z2 + (size_t)OFF[g] * 256, 256,
          dec_W1s + (size_t)g * 256 * 256, dec_W1n + (size_t)g * 256 * 256, dec_b1 + g * 256,
          dec_W2s + (size_t)g * 256 * 256, dec_W2n + (size_t)g * 256 * 256, dec_b2 + g * 256,
          Z512, 512, 256);
  fill_block<<<nb((long long)NA * 256), 256, 0, stream>>>(
      Z512 + (size_t)OFF[3] * 512, NA, 512, 256, 256, 0.f);

  // ----- 8) RGCN (l2) -> Z2, relu -----
  hconv(Z512, 512, 512, l2a_W, l2a_b, H, 256, 0);
  relu_k<<<nb((long long)NTOT * 256), 256, 0, stream>>>(H, (long long)NTOT * 256);
  hconv(H, 256, 256, l2b_W, l2b_b, Z2, 256, 0);
  relu_k<<<nb((long long)NTOT * 256), 256, 0, stream>>>(Z2, (long long)NTOT * 256);

  // ----- 9) RGCN (l3, Fout=2) + softmax -----
  hconv2(Z2, 256, 256, l3a_W, l3a_b, L1);
  relu_k<<<nb((long long)NTOT * 2), 256, 0, stream>>>(L1, (long long)NTOT * 2);
  hconv2(L1, 2, 2, l3b_W, l3b_b, L2);
  softmax2<<<nb(NTOT), 256, 0, stream>>>(L2, out, NTOT);
}
